// UpLayer_bak_22127671509191
// MI455X (gfx1250) — matrix operand order repaired and checked by reading, not compiled, not run on hardware
//
#include <hip/hip_runtime.h>
#include <hip/hip_bf16.h>

// ---------------------------------------------------------------------------
// Types
// ---------------------------------------------------------------------------
typedef __attribute__((ext_vector_type(16))) __bf16 v16bf;
typedef __attribute__((ext_vector_type(8)))  __bf16 v8bf;
typedef __attribute__((ext_vector_type(8)))  float  v8f;
typedef int v4i_vs __attribute__((vector_size(4 * sizeof(int))));
#define AS1 __attribute__((address_space(1)))
#define AS3 __attribute__((address_space(3)))

// Problem dims (fixed by reference)
#define BB 16
#define NP 512
#define NSD 256
#define CD 128
#define DA 64
#define KN 20
#define S2 (NP*KN)   /* 10240 */
#define NU (NP*2)    /* 1024  */

// CDNA5 async global->LDS path (guarded: fall back to load+ds_store if absent)
#if defined(__gfx1250__) && __has_builtin(__builtin_amdgcn_global_load_async_to_lds_b128)
#define USE_ASYNC_LDS 1
#else
#define USE_ASYNC_LDS 0
#endif

__device__ __forceinline__ __bf16 tobf(float f) {
    unsigned u = __builtin_bit_cast(unsigned, f);
    unsigned r = u + 0x7FFFu + ((u >> 16) & 1u);
    unsigned short h = (unsigned short)(r >> 16);
    return __builtin_bit_cast(__bf16, h);
}

// ---------------------------------------------------------------------------
// WMMA GEMM:  Y[b] (MxN f32) = A (MxKp bf16, shared) * X[b] (KpxN bf16) + bias
// Block tile 128x64, K-step 32, 8 waves, each wave 2x2 WMMA 16x16x32 bf16.
// A tile streamed with GLOBAL_LOAD_ASYNC_TO_LDS (ASYNCcnt) when available.
// ---------------------------------------------------------------------------
#define BM 128
#define BN 64
#define BK 32
#define LDA (BK + 8)

__device__ __forceinline__ v16bf ld_frag(const __bf16* p0, const __bf16* p1) {
    v8bf a = *(const v8bf*)p0;
    v8bf b = *(const v8bf*)p1;
    return __builtin_shufflevector(a, b, 0,1,2,3,4,5,6,7,8,9,10,11,12,13,14,15);
}

__global__ __launch_bounds__(256)
void k_gemm(const __bf16* __restrict__ A, const __bf16* __restrict__ Xall,
            const float* __restrict__ bias, float* __restrict__ Yall,
            int M, int N, int Kp, int relu) {
    __shared__ __align__(16) __bf16 sA[BM][LDA];
    __shared__ __align__(16) __bf16 sB[BN][LDA];   // transposed: [n][k]

    const int b  = blockIdx.z;
    const __bf16* X = Xall + (size_t)b * Kp * N;
    float*        Y = Yall + (size_t)b * M  * N;
    const int m0 = blockIdx.y * BM, n0 = blockIdx.x * BN;
    const int tid  = threadIdx.x;
    const int lane = tid & 31, wave = tid >> 5;
    const int wm = wave & 3, wn = wave >> 2;       // 4x2 wave grid -> 128x64
    const int lg = lane >> 4, l16 = lane & 15;

    v8f acc[2][2] = {};

    for (int k0 = 0; k0 < Kp; k0 += BK) {
        { // A tile: row-major, 16 bf16 per thread
            int r = tid >> 1, half = tid & 1;
            int gm = m0 + r;
            __bf16* dst = &sA[r][half * 16];
            if (gm < M) {
                const __bf16* src = A + (size_t)gm * Kp + k0 + half * 16;
#if USE_ASYNC_LDS
                __builtin_amdgcn_global_load_async_to_lds_b128(
                    (AS1 v4i_vs*)(v4i_vs*)(void*)(const void*)src,
                    (AS3 v4i_vs*)(v4i_vs*)(void*)dst, 0, 0);
#else
                v8bf v0 = *(const v8bf*)(src);
                v8bf v1 = *(const v8bf*)(src + 8);
                *(v8bf*)dst = v0; *(v8bf*)(dst + 8) = v1;
#endif
            } else {
                #pragma unroll
                for (int q = 0; q < 16; q++) dst[q] = tobf(0.f);
            }
        }
        { // X tile: load rows (contiguous in N), store transposed
            int kk = tid >> 3, ch = tid & 7;
            const __bf16* src = X + (size_t)(k0 + kk) * N + n0 + ch * 8;
            if (n0 + ch * 8 + 8 <= N) {
                v8bf v = *(const v8bf*)src;
                #pragma unroll
                for (int q = 0; q < 8; q++) sB[ch * 8 + q][kk] = v[q];
            } else {
                #pragma unroll
                for (int q = 0; q < 8; q++) {
                    int n = n0 + ch * 8 + q;
                    sB[ch * 8 + q][kk] = (n < N) ? src[q] : tobf(0.f);
                }
            }
        }
        // speculative prefetch of next K tile (global_prefetch_b8; OOB dropped)
        if (k0 + BK < Kp) {
            __builtin_prefetch(A + (size_t)(m0 + (tid >> 1)) * Kp + k0 + BK, 0, 1);
            __builtin_prefetch(X + (size_t)(k0 + BK + (tid >> 3)) * N + n0, 0, 1);
        }
#if USE_ASYNC_LDS
#if __has_builtin(__builtin_amdgcn_s_wait_asynccnt)
        __builtin_amdgcn_s_wait_asynccnt(0);
#else
        asm volatile("s_wait_asynccnt 0" ::: "memory");
#endif
#endif
        __syncthreads();

        v16bf af[2], bq[2];
        #pragma unroll
        for (int i = 0; i < 2; i++) {
            int r = wm * 32 + i * 16 + l16;
            af[i] = ld_frag(&sA[r][lg * 8], &sA[r][16 + lg * 8]);
        }
        #pragma unroll
        for (int j = 0; j < 2; j++) {
            int n = wn * 32 + j * 16 + l16;
            bq[j] = ld_frag(&sB[n][lg * 8], &sB[n][16 + lg * 8]);
        }
        #pragma unroll
        for (int i = 0; i < 2; i++)
            #pragma unroll
            for (int j = 0; j < 2; j++)
                acc[i][j] = __builtin_amdgcn_wmma_f32_16x16x32_bf16(
                    false, af[i], false, bq[j], (short)0, acc[i][j], false, false);
        __syncthreads();
    }

    #pragma unroll
    for (int i = 0; i < 2; i++)
        #pragma unroll
        for (int j = 0; j < 2; j++)
            #pragma unroll
            for (int r = 0; r < 8; r++) {
                int m = m0 + wm * 32 + i * 16 + lg * 8 + r;
                int n = n0 + wn * 32 + j * 16 + l16;
                if (m < M && n < N) {
                    float v = acc[i][j][r] + (bias ? bias[m] : 0.f);
                    if (relu) v = fmaxf(v, 0.f);
                    Y[(size_t)m * N + n] = v;
                }
            }
}

// ---------------------------------------------------------------------------
// Packing / elementwise / reduction kernels
// ---------------------------------------------------------------------------
__global__ void k_pack_w(const float* w, __bf16* o, int O, int C, int Kp) {
    int i = blockIdx.x * 256 + threadIdx.x;
    if (i >= O * Kp) return;
    int r = i / Kp, c = i % Kp;
    o[i] = (c < C) ? tobf(w[(size_t)r * C + c]) : tobf(0.f);
}

// ConvTranspose2d weight (C,O,2,1) -> row-major (O x Kp) for phase u
__global__ void k_pack_wT2(const float* w, __bf16* o, int C, int O, int u, int Kp) {
    int i = blockIdx.x * 256 + threadIdx.x;
    if (i >= O * Kp) return;
    int r = i / Kp, c = i % Kp;
    o[i] = (c < C) ? tobf(w[((size_t)c * O + r) * 2 + u]) : tobf(0.f);
}

// (B,C,S) f32 -> (B,Kp,S) bf16, zero-padded rows, optional relu
__global__ void k_pack_x(const float* x, __bf16* o, int C, int S, int Kp, int relu) {
    int b = blockIdx.z;
    size_t i = (size_t)blockIdx.x * 256 + threadIdx.x;
    if (i >= (size_t)Kp * S) return;
    int c = (int)(i / S); size_t s = i % S;
    float v = (c < C) ? x[((size_t)b * C + c) * S + s] : 0.f;
    if (relu) v = fmaxf(v, 0.f);
    o[(size_t)b * Kp * S + i] = tobf(v);
}

__global__ void k_pack_concat2(const float* a, const float* bsrc, __bf16* o,
                               int C1, int C2, int S) {
    int b = blockIdx.z;
    size_t i = (size_t)blockIdx.x * 256 + threadIdx.x;
    int Ct = C1 + C2;
    if (i >= (size_t)Ct * S) return;
    int c = (int)(i / S); size_t s = i % S;
    float v = (c < C1) ? a[((size_t)b * C1 + c) * S + s]
                       : bsrc[((size_t)b * C2 + (c - C1)) * S + s];
    o[(size_t)b * Ct * S + i] = tobf(v);
}

__global__ void k_pack_concat3m(const float* a, const float* mx, const float* fu,
                                __bf16* o, int S) {
    int b = blockIdx.z;
    size_t i = (size_t)blockIdx.x * 256 + threadIdx.x;
    if (i >= (size_t)384 * S) return;
    int c = (int)(i / S); size_t s = i % S;
    float v;
    if (c < 128)      v = a[((size_t)b * 128 + c) * S + s];
    else if (c < 256) v = mx[(size_t)b * 128 + (c - 128)];
    else              v = fu[((size_t)b * 128 + (c - 256)) * S + s];
    o[(size_t)b * 384 * S + i] = tobf(v);
}

__global__ void k_add(const float* a, const float* b, float* y, size_t n) {
    size_t i = (size_t)blockIdx.x * 256 + threadIdx.x;
    if (i < n) y[i] = a[i] + b[i];
}

__global__ void k_rep2(const float* x, float* o, int C, int S) {
    size_t i = (size_t)blockIdx.x * 256 + threadIdx.x;
    size_t tot = (size_t)BB * C * 2 * S;
    if (i >= tot) return;
    size_t j = i % (2 * S); size_t r = i / (2 * S);
    o[i] = x[r * S + (j >> 1)];
}

__global__ void k_add_rep2(const float* a, const float* ident, float* o, int C, int S) {
    size_t i = (size_t)blockIdx.x * 256 + threadIdx.x;
    size_t tot = (size_t)BB * C * 2 * S;
    if (i >= tot) return;
    size_t j = i % (2 * S); size_t r = i / (2 * S);
    o[i] = a[i] + ident[r * S + (j >> 1)];
}

// BatchNorm2d training-mode stats: per channel over (B, spatial)
__global__ void k_bn_stats(const float* x, const float* g, const float* bt,
                           float* scale, float* shift, int C, size_t S, int Bn) {
    int c = blockIdx.x, t = threadIdx.x;
    float s = 0.f, s2 = 0.f;
    for (int b = 0; b < Bn; b++) {
        const float* p = x + ((size_t)b * C + c) * S;
        for (size_t i = t; i < S; i += 256) { float v = p[i]; s += v; s2 += v * v; }
    }
    __shared__ float rs[256], rs2[256];
    rs[t] = s; rs2[t] = s2; __syncthreads();
    for (int o = 128; o > 0; o >>= 1) {
        if (t < o) { rs[t] += rs[t + o]; rs2[t] += rs2[t + o]; }
        __syncthreads();
    }
    if (t == 0) {
        float n = (float)S * (float)Bn;
        float m = rs[0] / n;
        float v = rs2[0] / n - m * m;
        float sc = g[c] * rsqrtf(v + 1e-5f);
        scale[c] = sc; shift[c] = bt[c] - m * sc;
    }
}

__global__ void k_bn_relu_pack(const float* x, const float* sc, const float* sh,
                               __bf16* o, int C, int S) {
    int b = blockIdx.z;
    size_t i = (size_t)blockIdx.x * 256 + threadIdx.x;
    if (i >= (size_t)C * S) return;
    int c = (int)(i / S);
    float v = x[(size_t)b * C * S + i] * sc[c] + sh[c];
    o[(size_t)b * C * S + i] = tobf(fmaxf(v, 0.f));
}

__global__ void k_knn(const float* pos, int* idx) {
    int i = blockIdx.x * 256 + threadIdx.x;
    if (i >= BB * NP) return;
    int b = i / NP, n = i % NP;
    const float* P = pos + (size_t)b * 3 * NP;
    float x = P[n], y = P[NP + n], z = P[2 * NP + n];
    float bd[KN]; int bi[KN];
    for (int j = 0; j < KN; j++) { bd[j] = 3.4e38f; bi[j] = 0; }
    for (int m = 0; m < NP; m++) {
        float dx = P[m] - x, dy = P[NP + m] - y, dz = P[2 * NP + m] - z;
        float d = dx * dx + dy * dy + dz * dz;
        if (d < bd[KN - 1]) {
            int j = KN - 1;
            while (j > 0 && bd[j - 1] > d) { bd[j] = bd[j - 1]; bi[j] = bi[j - 1]; j--; }
            bd[j] = d; bi[j] = m;
        }
    }
    for (int j = 0; j < KN; j++) idx[((size_t)b * NP + n) * KN + j] = bi[j];
}

// out[b,c,n,k] = fc[b,c,n] - fg[b,c,idx[b,n,k]]
__global__ void k_rel_group2(const float* fc, const float* fg, const int* idx,
                             float* o, int C) {
    size_t tot = (size_t)BB * C * NP * KN;
    size_t i = (size_t)blockIdx.x * 256 + threadIdx.x;
    if (i >= tot) return;
    int kk = (int)(i % KN); size_t r = i / KN;
    int n = (int)(r % NP);  size_t r2 = r / NP;
    int c = (int)(r2 % C);  int b = (int)(r2 / C);
    int id = idx[((size_t)b * NP + n) * KN + kk];
    size_t rowb = ((size_t)b * C + c) * NP;
    o[i] = fc[rowb + n] - fg[rowb + id];
}

// pack (qk+pe) transposed to channels=N layout: out (B, NP, DA*KN) bf16
__global__ void k_addpack_xN(const float* qk, const float* pe, __bf16* o) {
    size_t tot = (size_t)BB * NP * DA * KN;
    size_t i = (size_t)blockIdx.x * 256 + threadIdx.x;
    if (i >= tot) return;
    int j = (int)(i % (DA * KN)); size_t r = i / (DA * KN);
    int n = (int)(r % NP); int b = (int)(r / NP);
    int c = j / KN, kk = j % KN;
    size_t src = (((size_t)b * DA + c) * NP + n) * KN + kk;
    o[i] = tobf(qk[src] + pe[src]);
}

__global__ void k_softmax20(float* x, size_t rows) {
    size_t r = (size_t)blockIdx.x * 256 + threadIdx.x;
    if (r >= rows) return;
    float* p = x + r * KN;
    float m = -3.4e38f;
    for (int j = 0; j < KN; j++) m = fmaxf(m, p[j]);
    float s = 0.f;
    for (int j = 0; j < KN; j++) { float e = __expf(p[j] - m); p[j] = e; s += e; }
    float inv = 1.f / s;
    for (int j = 0; j < KN; j++) p[j] *= inv;
}

// agg[b,c,n] = sum_k attN[b,n,c*20+k] * (value[b,c,n] + pe[b,c,n,k])
__global__ void k_aggN(const float* att, const float* val, const float* pe, float* agg) {
    size_t tot = (size_t)BB * DA * NP;
    size_t i = (size_t)blockIdx.x * 256 + threadIdx.x;
    if (i >= tot) return;
    int n = (int)(i % NP); size_t r = i / NP;
    int c = (int)(r % DA); int b = (int)(r / DA);
    float v0 = val[((size_t)b * DA + c) * NP + n];
    const float* pa = att + ((size_t)b * NP + n) * (DA * KN) + c * KN;
    const float* pp = pe + ((((size_t)b * DA + c) * NP) + n) * KN;
    float s = 0.f;
    for (int kk = 0; kk < KN; kk++) s += pa[kk] * (v0 + pp[kk]);
    agg[i] = s;
}

// am1 input = qk_rel + y_N(reshape of end1) + pos_emb + upfeat_rel, packed bf16
__global__ void k_build_am1in(const float* qk, const float* e1, const float* pe,
                              const float* ur, __bf16* o) {
    size_t tot = (size_t)BB * DA * S2;
    size_t i = (size_t)blockIdx.x * 256 + threadIdx.x;
    if (i >= tot) return;
    size_t s = i % S2; size_t r = i / S2;
    int c = (int)(r % DA); int b = (int)(r / DA);
    size_t flat = (size_t)c * S2 + s;          // torch reshape (B,-1,N,k)
    int c2 = (int)(flat / NP), ns = (int)(flat % NP);
    float yN = e1[((size_t)b * (DA * KN) + c2) * NP + ns];
    o[i] = tobf(qk[i] + yN + pe[i] + ur[i]);
}

// val[b,c,n,k] = value[b,c,idx] + pe + ur
__global__ void k_buildval(const float* val, const float* pe, const float* ur,
                           const int* idx, float* o) {
    size_t tot = (size_t)BB * DA * S2;
    size_t i = (size_t)blockIdx.x * 256 + threadIdx.x;
    if (i >= tot) return;
    size_t s = i % S2; size_t r = i / S2;
    int c = (int)(r % DA); int b = (int)(r / DA);
    int n = (int)(s / KN), kk = (int)(s % KN);
    int id = idx[((size_t)b * NP + n) * KN + kk];
    o[i] = val[((size_t)b * DA + c) * NP + id] + pe[i] + ur[i];
}

__global__ void k_agg(const float* att, const float* val, float* agg) {
    size_t tot = (size_t)BB * DA * NP;
    size_t i = (size_t)blockIdx.x * 256 + threadIdx.x;
    if (i >= tot) return;
    size_t base = i * KN;
    float s = 0.f;
    for (int kk = 0; kk < KN; kk++) s += att[base + kk] * val[base + kk];
    agg[i] = s;
}

__global__ void k_agg_up(const float* att0, const float* att1, const float* val,
                         float* agg) {
    size_t tot = (size_t)BB * DA * NU;
    size_t i = (size_t)blockIdx.x * 256 + threadIdx.x;
    if (i >= tot) return;
    int j = (int)(i % NU); size_t r = i / NU;
    int n = j >> 1, u = j & 1;
    size_t base = (r * NP + n) * KN;
    const float* A = u ? att1 : att0;
    float s = 0.f;
    for (int kk = 0; kk < KN; kk++) s += A[base + kk] * val[base + kk];
    agg[i] = s;
}

__global__ void k_interp_idx(const float* pcd, const float* seed, int* idx3, float* w3) {
    int i = blockIdx.x * 256 + threadIdx.x;
    if (i >= BB * NP) return;
    int b = i / NP, n = i % NP;
    const float* P = pcd + (size_t)b * 3 * NP;
    const float* Sd = seed + (size_t)b * 3 * NSD;
    float x = P[n], y = P[NP + n], z = P[2 * NP + n];
    float bd[3] = {3.4e38f, 3.4e38f, 3.4e38f}; int bi[3] = {0, 0, 0};
    for (int m = 0; m < NSD; m++) {
        float dx = Sd[m] - x, dy = Sd[NSD + m] - y, dz = Sd[2 * NSD + m] - z;
        float d = dx * dx + dy * dy + dz * dz;
        if (d < bd[2]) {
            int j = 2;
            while (j > 0 && bd[j - 1] > d) { bd[j] = bd[j - 1]; bi[j] = bi[j - 1]; j--; }
            bd[j] = d; bi[j] = m;
        }
    }
    float w0 = 1.f / (bd[0] + 1e-8f), w1 = 1.f / (bd[1] + 1e-8f), w2 = 1.f / (bd[2] + 1e-8f);
    float s = w0 + w1 + w2;
    size_t o = (size_t)i * 3;
    idx3[o] = bi[0]; idx3[o + 1] = bi[1]; idx3[o + 2] = bi[2];
    w3[o] = w0 / s;  w3[o + 1] = w1 / s;  w3[o + 2] = w2 / s;
}

__global__ void k_interp_gather(const float* seedf, const int* idx3, const float* w3,
                                float* fup) {
    size_t tot = (size_t)BB * CD * NP;
    size_t i = (size_t)blockIdx.x * 256 + threadIdx.x;
    if (i >= tot) return;
    int n = (int)(i % NP); size_t r = i / NP;
    int c = (int)(r % CD); int b = (int)(r / CD);
    size_t o = ((size_t)b * NP + n) * 3;
    const float* F = seedf + ((size_t)b * CD + c) * NSD;
    fup[i] = w3[o] * F[idx3[o]] + w3[o + 1] * F[idx3[o + 1]] + w3[o + 2] * F[idx3[o + 2]];
}

__global__ void k_rowmax(const float* x, float* mx, int C, int S) {
    int i = blockIdx.x * 256 + threadIdx.x;
    if (i >= BB * C) return;
    const float* p = x + (size_t)i * S;
    float m = -3.4e38f;
    for (int s = 0; s < S; s++) m = fmaxf(m, p[s]);
    mx[i] = m;
}

__global__ void k_tanh_out(const float* d2, const float* pcd, float* out) {
    size_t tot = (size_t)BB * 3 * NU;
    size_t i = (size_t)blockIdx.x * 256 + threadIdx.x;
    if (i >= tot) return;
    int j = (int)(i % NU); size_t r = i / NU;
    out[i] = pcd[r * NP + (j >> 1)] + tanhf(d2[i]);
}

// ---------------------------------------------------------------------------
// Host orchestration
// ---------------------------------------------------------------------------
static inline unsigned g1u(size_t n) { return (unsigned)((n + 255) / 256); }

struct Bump {
    char* p; size_t o;
    void* take(size_t b) { size_t a = (o + 255) & ~(size_t)255; o = a + b; return p + a; }
    float*  f(size_t n) { return (float*)take(n * 4); }
    __bf16* h(size_t n) { return (__bf16*)take(n * 2); }
    int*    i(size_t n) { return (int*)take(n * 4); }
};

static void gemm(const __bf16* A, const __bf16* X, const float* bias, float* Y,
                 int M, int N, int Kp, int relu, hipStream_t st) {
    dim3 grid((N + BN - 1) / BN, (M + BM - 1) / BM, BB);
    k_gemm<<<grid, 256, 0, st>>>(A, X, bias, Y, M, N, Kp, relu);
}

static __bf16* packw(Bump& al, const float* w, int O, int C, hipStream_t st) {
    int Kp = ((C + 31) / 32) * 32;
    __bf16* o = al.h((size_t)O * Kp);
    k_pack_w<<<g1u((size_t)O * Kp), 256, 0, st>>>(w, o, O, C, Kp);
    return o;
}

static __bf16* packx(Bump& al, const float* x, int C, int S, int relu, hipStream_t st) {
    int Kp = ((C + 31) / 32) * 32;
    __bf16* o = al.h((size_t)BB * Kp * S);
    dim3 grid(g1u((size_t)Kp * S), 1, BB);
    k_pack_x<<<grid, 256, 0, st>>>(x, o, C, S, Kp, relu);
    return o;
}

static __bf16* bn_relu_pack(Bump& al, const float* x, const float* g, const float* b,
                            int C, int S, hipStream_t st) {
    float* scale = al.f(C); float* shift = al.f(C);
    k_bn_stats<<<C, 256, 0, st>>>(x, g, b, scale, shift, C, (size_t)S, BB);
    __bf16* o = al.h((size_t)BB * C * S);
    dim3 grid(g1u((size_t)C * S), 1, BB);
    k_bn_relu_pack<<<grid, 256, 0, st>>>(x, scale, shift, o, C, S);
    return o;
}

// param index layout: sorted-key pytree flatten after 4 positional inputs
enum { IN_PCD = 0, IN_SEED, IN_SEEDF, IN_KPREV,
       P_MD_C1_B = 4, P_MD_C1_W, P_MD_C2_B, P_MD_C2_W,
       P_MDF_C1_B = 8, P_MDF_C1_W, P_MDF_C2_B, P_MDF_C2_W, P_MDF_SC_B, P_MDF_SC_W,
       P_M1_C1_B = 14, P_M1_C1_W, P_M1_C2_B, P_M1_C2_W,
       P_M2_C1_B = 18, P_M2_C1_W, P_M2_C2_B, P_M2_C2_W,
       P_T1 = 22, P_T2 = 58 };
enum { T_AM1_B = 0, T_AM1_W, T_AM2_B, T_AM2_W, T_AMN1_B, T_AMN1_W, T_AMN2_B, T_AMN2_W,
       T_AMNBN_B, T_AMNBN_G, T_AMBN_B, T_AMBN_G, T_END_B, T_END_W, T_END1_B, T_END1_W,
       T_KEY_B, T_KEY_W, T_V_C1_B, T_V_C1_W, T_V_C2_B, T_V_C2_W, T_V_SC_B, T_V_SC_W,
       T_POS1_B, T_POS1_W, T_POS2_B, T_POS2_W, T_POSBN_B, T_POSBN_G,
       T_Q_B, T_Q_W, T_UPF_B, T_UPF_W, T_VAL_B, T_VAL_W };

static float* transformer(hipStream_t st, const float* const* P, Bump& al,
                          const __bf16* keyPk, const float* keyF, const float* queryF,
                          const __bf16* fupPk, const int* idx, const __bf16* posRelPk,
                          int up) {
    const int Sout = up ? NU : NP;
    float* out = al.f((size_t)BB * CD * Sout);
    size_t ck = al.o;

    // ---- mlp_v (MLP_Res) on concat[key_f, query_f] ----
    __bf16* pc = al.h((size_t)BB * 256 * NP);
    { dim3 g(g1u((size_t)256 * NP), 1, BB);
      k_pack_concat2<<<g, 256, 0, st>>>(keyF, queryF, pc, CD, CD, NP); }
    float* v1 = al.f((size_t)BB * CD * NP);
    gemm(packw(al, P[T_V_C1_W], 128, 256, st), pc, P[T_V_C1_B], v1, 128, NP, 256, 0, st);
    __bf16* v1p = packx(al, v1, 128, NP, 1, st);
    float* v2 = al.f((size_t)BB * CD * NP);
    gemm(packw(al, P[T_V_C2_W], 128, 128, st), v1p, P[T_V_C2_B], v2, 128, NP, 128, 0, st);
    gemm(packw(al, P[T_V_SC_W], 128, 256, st), pc, P[T_V_SC_B], v1, 128, NP, 256, 0, st);
    float* vfull = al.f((size_t)BB * CD * NP);   // identity
    k_add<<<g1u((size_t)BB * CD * NP), 256, 0, st>>>(v2, v1, vfull, (size_t)BB * CD * NP);
    __bf16* vfp = packx(al, vfull, 128, NP, 0, st);

    // ---- attention projections ----
    float* k_ = al.f((size_t)BB * DA * NP);
    gemm(packw(al, P[T_KEY_W], 64, 128, st), keyPk, P[T_KEY_B], k_, 64, NP, 128, 0, st);
    __bf16* qPk = packx(al, queryF, 128, NP, 0, st);
    float* q_ = al.f((size_t)BB * DA * NP);
    gemm(packw(al, P[T_Q_W], 64, 128, st), qPk, P[T_Q_B], q_, 64, NP, 128, 0, st);
    float* vat = al.f((size_t)BB * DA * NP);
    gemm(packw(al, P[T_VAL_W], 64, 128, st), vfp, P[T_VAL_B], vat, 64, NP, 128, 0, st);
    float* uf = al.f((size_t)BB * DA * NP);
    gemm(packw(al, P[T_UPF_W], 64, 128, st), fupPk, P[T_UPF_B], uf, 64, NP, 128, 0, st);

    float* qk = al.f((size_t)BB * DA * S2);
    k_rel_group2<<<g1u((size_t)BB * DA * S2), 256, 0, st>>>(q_, k_, idx, qk, DA);
    float* ur = al.f((size_t)BB * DA * S2);
    k_rel_group2<<<g1u((size_t)BB * DA * S2), 256, 0, st>>>(uf, uf, idx, ur, DA);

    // ---- positional embedding ----
    float* pe0 = al.f((size_t)BB * DA * S2);
    gemm(packw(al, P[T_POS1_W], 64, 3, st), posRelPk, P[T_POS1_B], pe0, 64, S2, 32, 0, st);
    __bf16* pe0p = bn_relu_pack(al, pe0, P[T_POSBN_G], P[T_POSBN_B], 64, S2, st);
    float* pe = al.f((size_t)BB * DA * S2);
    gemm(packw(al, P[T_POS2_W], 64, 64, st), pe0p, P[T_POS2_B], pe, 64, S2, 64, 0, st);

    // ---- amN branch (channels = N) ----
    float* aggn = al.f((size_t)BB * DA * NP);
    float* e1   = al.f((size_t)BB * (DA * KN) * NP);
    {
        size_t ck2 = al.o;
        __bf16* xN = al.h((size_t)BB * NP * DA * KN);
        k_addpack_xN<<<g1u((size_t)BB * NP * DA * KN), 256, 0, st>>>(qk, pe, xN);
        float* aN1 = al.f((size_t)BB * 2048 * (DA * KN));
        gemm(packw(al, P[T_AMN1_W], 2048, 512, st), xN, P[T_AMN1_B], aN1,
             2048, DA * KN, 512, 0, st);
        __bf16* aN1p = bn_relu_pack(al, aN1, P[T_AMNBN_G], P[T_AMNBN_B], 2048, DA * KN, st);
        float* attN = al.f((size_t)BB * 512 * (DA * KN));
        gemm(packw(al, P[T_AMN2_W], 512, 2048, st), aN1p, P[T_AMN2_B], attN,
             512, DA * KN, 2048, 0, st);
        k_softmax20<<<g1u((size_t)BB * 512 * DA), 256, 0, st>>>(attN, (size_t)BB * 512 * DA);
        k_aggN<<<g1u((size_t)BB * DA * NP), 256, 0, st>>>(attN, vat, pe, aggn);
        al.o = ck2;  // release amN temporaries
    }
    __bf16* aggp = packx(al, aggn, 64, NP, 0, st);
    gemm(packw(al, P[T_END1_W], DA * KN, 64, st), aggp, P[T_END1_B], e1,
         DA * KN, NP, 64, 0, st);

    // ---- main attention ----
    __bf16* am1in = al.h((size_t)BB * DA * S2);
    k_build_am1in<<<g1u((size_t)BB * DA * S2), 256, 0, st>>>(qk, e1, pe, ur, am1in);
    float* hpre = al.f((size_t)BB * 256 * S2);
    gemm(packw(al, P[T_AM1_W], 256, 64, st), am1in, P[T_AM1_B], hpre, 256, S2, 64, 0, st);
    __bf16* hp = bn_relu_pack(al, hpre, P[T_AMBN_G], P[T_AMBN_B], 256, S2, st);
    float* val = al.f((size_t)BB * DA * S2);
    k_buildval<<<g1u((size_t)BB * DA * S2), 256, 0, st>>>(vat, pe, ur, idx, val);

    if (!up) {
        float* att = hpre;  // reuse (hpre no longer needed)
        gemm(packw(al, P[T_AM2_W], 64, 256, st), hp, P[T_AM2_B], att, 64, S2, 256, 0, st);
        k_softmax20<<<g1u((size_t)BB * DA * NP), 256, 0, st>>>(att, (size_t)BB * DA * NP);
        float* agg = aggn;  // reuse
        k_agg<<<g1u((size_t)BB * DA * NP), 256, 0, st>>>(att, val, agg);
        __bf16* ap = packx(al, agg, 64, NP, 0, st);
        float* eo = al.f((size_t)BB * CD * NP);
        gemm(packw(al, P[T_END_W], 128, 64, st), ap, P[T_END_B], eo, 128, NP, 64, 0, st);
        k_add<<<g1u((size_t)BB * CD * NP), 256, 0, st>>>(eo, vfull, out, (size_t)BB * CD * NP);
    } else {
        __bf16* w0 = al.h((size_t)64 * 256);
        __bf16* w1 = al.h((size_t)64 * 256);
        k_pack_wT2<<<g1u((size_t)64 * 256), 256, 0, st>>>(P[T_AM2_W], w0, 256, 64, 0, 256);
        k_pack_wT2<<<g1u((size_t)64 * 256), 256, 0, st>>>(P[T_AM2_W], w1, 256, 64, 1, 256);
        float* att0 = hpre;  // reuse
        float* att1 = al.f((size_t)BB * DA * S2);
        gemm(w0, hp, P[T_AM2_B], att0, 64, S2, 256, 0, st);
        gemm(w1, hp, P[T_AM2_B], att1, 64, S2, 256, 0, st);
        k_softmax20<<<g1u((size_t)BB * DA * NP), 256, 0, st>>>(att0, (size_t)BB * DA * NP);
        k_softmax20<<<g1u((size_t)BB * DA * NP), 256, 0, st>>>(att1, (size_t)BB * DA * NP);
        float* aggu = al.f((size_t)BB * DA * NU);
        k_agg_up<<<g1u((size_t)BB * DA * NU), 256, 0, st>>>(att0, att1, val, aggu);
        __bf16* ap = packx(al, aggu, 64, NU, 0, st);
        float* eo = al.f((size_t)BB * CD * NU);
        gemm(packw(al, P[T_END_W], 128, 64, st), ap, P[T_END_B], eo, 128, NU, 64, 0, st);
        k_add_rep2<<<g1u((size_t)BB * CD * NU), 256, 0, st>>>(eo, vfull, out, CD, NP);
    }
    al.o = ck;
    return out;
}

extern "C" void kernel_launch(void* const* d_in, const int* in_sizes, int n_in,
                              void* d_out, int out_size, void* d_ws, size_t ws_size,
                              hipStream_t stream) {
    (void)in_sizes; (void)n_in; (void)out_size; (void)ws_size;
    hipStream_t st = stream;
    Bump al{(char*)d_ws, 0};
    const float* pcd   = (const float*)d_in[IN_PCD];
    const float* seed  = (const float*)d_in[IN_SEED];
    const float* seedf = (const float*)d_in[IN_SEEDF];
    const float* Kprev = (const float*)d_in[IN_KPREV];
    float* outPcd = (float*)d_out;
    float* outK   = (float*)d_out + (size_t)BB * 3 * NU;

    // ---- three-NN interpolation of seed features ----
    int*   idx3 = al.i((size_t)BB * NP * 3);
    float* w3   = al.f((size_t)BB * NP * 3);
    k_interp_idx<<<g1u((size_t)BB * NP), 256, 0, st>>>(pcd, seed, idx3, w3);
    float* fup = al.f((size_t)BB * CD * NP);
    k_interp_gather<<<g1u((size_t)BB * CD * NP), 256, 0, st>>>(seedf, idx3, w3, fup);

    // ---- mlp1 ----
    __bf16* pcdP = packx(al, pcd, 3, NP, 0, st);
    float* f1 = al.f((size_t)BB * 64 * NP);
    gemm(packw(al, (const float*)d_in[P_M1_C1_W], 64, 3, st), pcdP,
         (const float*)d_in[P_M1_C1_B], f1, 64, NP, 32, 0, st);
    __bf16* f1p = packx(al, f1, 64, NP, 1, st);
    float* m1o = al.f((size_t)BB * 128 * NP);
    gemm(packw(al, (const float*)d_in[P_M1_C2_W], 128, 64, st), f1p,
         (const float*)d_in[P_M1_C2_B], m1o, 128, NP, 64, 0, st);
    float* mx = al.f((size_t)BB * 128);
    k_rowmax<<<g1u((size_t)BB * 128), 256, 0, st>>>(m1o, mx, 128, NP);

    // ---- mlp2 on concat[feat_1, max, feat_up] ----
    __bf16* cat = al.h((size_t)BB * 384 * NP);
    { dim3 g(g1u((size_t)384 * NP), 1, BB);
      k_pack_concat3m<<<g, 256, 0, st>>>(m1o, mx, fup, cat, NP); }
    float* t0 = al.f((size_t)BB * 256 * NP);
    gemm(packw(al, (const float*)d_in[P_M2_C1_W], 256, 384, st), cat,
         (const float*)d_in[P_M2_C1_B], t0, 256, NP, 384, 0, st);
    __bf16* t0p = packx(al, t0, 256, NP, 1, st);
    float* Q = al.f((size_t)BB * CD * NP);
    gemm(packw(al, (const float*)d_in[P_M2_C2_W], 128, 256, st), t0p,
         (const float*)d_in[P_M2_C2_B], Q, 128, NP, 256, 0, st);

    // ---- shared KNN / packs ----
    int* idx = al.i((size_t)BB * NP * KN);
    k_knn<<<g1u((size_t)BB * NP), 256, 0, st>>>(pcd, idx);
    float* posRel = al.f((size_t)BB * 3 * S2);
    k_rel_group2<<<g1u((size_t)BB * 3 * S2), 256, 0, st>>>(pcd, pcd, idx, posRel, 3);
    __bf16* posRelP = packx(al, posRel, 3, S2, 0, st);
    __bf16* KprevP  = packx(al, Kprev, 128, NP, 0, st);
    __bf16* fupP    = packx(al, fup, 128, NP, 0, st);

    // ---- two UpTransformers ----
    float* H  = transformer(st, (const float* const*)(d_in + P_T1), al,
                            KprevP, Kprev, Q, fupP, idx, posRelP, 0);
    float* FC = transformer(st, (const float* const*)(d_in + P_T2), al,
                            KprevP, Kprev, H, fupP, idx, posRelP, 1);

    // ---- tail: K_curr = MLP_Res(mdf, concat[feat_child, H_up]) ----
    float* Hup = al.f((size_t)BB * CD * NU);
    k_rep2<<<g1u((size_t)BB * CD * NU), 256, 0, st>>>(H, Hup, CD, NP);
    __bf16* cat2 = al.h((size_t)BB * 256 * NU);
    { dim3 g(g1u((size_t)256 * NU), 1, BB);
      k_pack_concat2<<<g, 256, 0, st>>>(FC, Hup, cat2, CD, CD, NU); }
    float* u1 = al.f((size_t)BB * CD * NU);
    gemm(packw(al, (const float*)d_in[P_MDF_C1_W], 128, 256, st), cat2,
         (const float*)d_in[P_MDF_C1_B], u1, 128, NU, 256, 0, st);
    __bf16* u1p = packx(al, u1, 128, NU, 1, st);
    float* u2 = al.f((size_t)BB * CD * NU);
    gemm(packw(al, (const float*)d_in[P_MDF_C2_W], 128, 128, st), u1p,
         (const float*)d_in[P_MDF_C2_B], u2, 128, NU, 128, 0, st);
    gemm(packw(al, (const float*)d_in[P_MDF_SC_W], 128, 256, st), cat2,
         (const float*)d_in[P_MDF_SC_B], u1, 128, NU, 256, 0, st);
    k_add<<<g1u((size_t)BB * CD * NU), 256, 0, st>>>(u2, u1, outK, (size_t)BB * CD * NU);

    // ---- displacement head: delta = tanh(md(relu(K_curr))) ----
    __bf16* kcp = packx(al, outK, 128, NU, 1, st);
    float* d1 = u2;  // reuse
    gemm(packw(al, (const float*)d_in[P_MD_C1_W], 64, 128, st), kcp,
         (const float*)d_in[P_MD_C1_B], d1, 64, NU, 128, 0, st);
    __bf16* d1p = packx(al, d1, 64, NU, 1, st);
    float* d2 = al.f((size_t)BB * 3 * NU);
    gemm(packw(al, (const float*)d_in[P_MD_C2_W], 3, 64, st), d1p,
         (const float*)d_in[P_MD_C2_B], d2, 3, NU, 64, 0, st);
    k_tanh_out<<<g1u((size_t)BB * 3 * NU), 256, 0, st>>>(d2, pcd, outPcd);
}
